// GraphAttentionLayer_55602646614231
// MI455X (gfx1250) — matrix operand order repaired and checked by reading, not compiled, not run on hardware
//
#include <hip/hip_runtime.h>
#include <hip/hip_bf16.h>

// GAT fused layer for MI455X (gfx1250), wave32, WMMA bf16 path.
// B=4096, N=200, D_IN=D_OUT=128. One workgroup (8 waves) per batch element.
// GEMM (h @ W^T) in bf16 WMMA w/ f32 accumulate; softmax + aggregation fused in LDS.
// - pair of 16-row tiles per iteration -> two independent WMMA chains (no hazard nops)
// - B (W) fragments hoisted into registers: loop-invariant, saves 1/3 of LDS loads
// - distance-2 software pipeline: HBM loads for pair p+2 issue in iteration p

#define BATCH   4096
#define NNODE   200
#define DIN     128
#define DOUT    128
#define NEGSL   0.2f
#define PTILES  7       // ceil(200/32) pair-iterations, 32 rows each
#define WHS     129     // padded f32 row stride for Wh in LDS (odd -> conflict-free)

typedef __bf16 v16bf __attribute__((ext_vector_type(16)));
typedef __bf16 v2bf  __attribute__((ext_vector_type(2)));
typedef float  v8f   __attribute__((ext_vector_type(8)));
typedef float  v2f   __attribute__((ext_vector_type(2)));

union Frag { unsigned int u[8]; v16bf v; };

// f32 pair -> packed bf16 dword: one v_cvt_pk_bf16_f32
__device__ __forceinline__ unsigned int pk_bf16(float a, float b) {
  union { v2bf v; unsigned int u; } cv;
  v2f in = {a, b};
  cv.v = __builtin_convertvector(in, v2bf);
  return cv.u;
}

__device__ __forceinline__ float wave_max32(float v) {
  #pragma unroll
  for (int m = 16; m > 0; m >>= 1) v = fmaxf(v, __shfl_xor(v, m, 32));
  return v;
}
__device__ __forceinline__ float wave_sum32(float v) {
  #pragma unroll
  for (int m = 16; m > 0; m >>= 1) v += __shfl_xor(v, m, 32);
  return v;
}

__global__ __launch_bounds__(256)
void gat_fused_kernel(const float* __restrict__ hsrc,
                      const int*   __restrict__ adj,
                      const float* __restrict__ W,
                      const float* __restrict__ avec,
                      float*       __restrict__ out) {
  // LDS: 32K (W bf16) + 16K (32x128 A tile x2) + 103.2K (Wh f32) + 3K misc ~= 152 KB
  __shared__ unsigned int Wu[DOUT * (DIN / 2)];   // bf16x2 packed, [o][f/2] row-major
  __shared__ unsigned int Au[2][32 * (DIN / 2)];  // double-buffered 32x128 A tile (bf16x2)
  __shared__ float Whf[NNODE * WHS];              // f32 result of h[b] @ W^T
  __shared__ float aLds[2 * DOUT];                // a1 | a2
  __shared__ float rbuf[256];                     // reduction / partial-sum scratch
  __shared__ float ebuf[256];                     // final attention weights

  const int t    = threadIdx.x;
  const int b    = blockIdx.x;
  const int wave = t >> 5;       // 8 waves; wave == output column tile (16 cols each)
  const int lane = t & 31;
  const int hh   = lane >> 4;    // half-wave selector (ISA fragment layout)
  const int l15  = lane & 15;

  const float4* H4 = (const float4*)(hsrc + (size_t)b * NNODE * DIN);

  // issue global loads of a 32x128 pair-tile into registers (no wait yet)
  auto loadP = [&](int p, float4 q[4]) {
    #pragma unroll
    for (int i = 0; i < 4; ++i) {
      int j = t + 256 * i;                 // 1024 float4 per 32x128 tile
      int g = p * 32 + (j >> 5);           // 32 float4 per 128-wide row
      q[i] = make_float4(0.f, 0.f, 0.f, 0.f);
      if (g < NNODE) q[i] = H4[(size_t)g * 32 + (j & 31)];
    }
  };
  // pack to bf16 and commit to LDS (loadcnt wait lands here, ~1 iteration later)
  auto commitP = [&](int buf, const float4 q[4]) {
    #pragma unroll
    for (int i = 0; i < 4; ++i) {
      int j = t + 256 * i;
      Au[buf][2 * j]     = pk_bf16(q[i].x, q[i].y);
      Au[buf][2 * j + 1] = pk_bf16(q[i].z, q[i].w);
    }
  };
  // prefetch further ahead into cache (global_prefetch_b8)
  auto prefP = [&](int p) {
    int g = p * 32 + (t >> 3);
    if (g < NNODE) __builtin_prefetch(&H4[(size_t)g * 32 + 4 * (t & 7)], 0, 1);
  };

  // ---------------- stage W (f32 -> packed bf16) and a ----------------
  {
    const float4* W4 = (const float4*)W;          // 4096 float4
    #pragma unroll
    for (int i = 0; i < 16; ++i) {
      int j = t + 256 * i;
      float4 w = W4[j];
      Wu[2 * j]     = pk_bf16(w.x, w.y);
      Wu[2 * j + 1] = pk_bf16(w.z, w.w);
    }
    aLds[t] = avec[t];
  }

  // pipeline prologue: pair0 committed, pair1 loads in flight
  float4 ca[4];   // registers holding pair (p+1), committed during iteration p
  float4 lb[4];   // registers receiving pair (p+2), loaded during iteration p
  loadP(0, ca);
  commitP(0, ca);
  loadP(1, ca);
  prefP(2);
  __syncthreads();

  // hoist loop-invariant B (W) fragments into registers: 4 k-steps x 8 dwords.
  // B 32x16 bf16 layout: col = lane&15; dword v holds K-pair 16*(v>>2) + 8*half + 2*(v&3).
  // B(k,n) = W[o = wave*16+n][f = kk*32+k]   (Wh = h @ W^T)
  Frag Bk[4];
  #pragma unroll
  for (int kk = 0; kk < 4; ++kk)
    #pragma unroll
    for (int v = 0; v < 8; ++v)
      Bk[kk].u[v] = Wu[(wave * 16 + l15) * 64 + kk * 16 + 8 * (v >> 2) + 4 * hh + (v & 3)];

  // ---------------- GEMM: Wh = h[b] (200x128) @ W^T (128x128) ----------------
  for (int p = 0; p < PTILES; ++p) {
    const int cur = p & 1;
    if (p + 2 < PTILES) loadP(p + 2, lb);        // issue HBM loads 2 pairs ahead
    if (p + 3 < PTILES) prefP(p + 3);

    v8f acc0 = {0.f, 0.f, 0.f, 0.f, 0.f, 0.f, 0.f, 0.f};
    v8f acc1 = acc0;
    #pragma unroll
    for (int kk = 0; kk < 4; ++kk) {             // K = 128 in 4 steps of 32
      Frag A0, A1;
      #pragma unroll
      for (int v = 0; v < 8; ++v) {
        // A 16x32 bf16 layout (ISA 7.12.2): row = lane&15; dword v holds K-pair
        // starting at 16*(v>>2) + 8*half + 2*(v&3)
        int g = v >> 2, vv = v & 3;
        A0.u[v] = Au[cur][l15 * 64        + kk * 16 + 8 * g + 4 * hh + vv];
        A1.u[v] = Au[cur][(16 + l15) * 64 + kk * 16 + 8 * g + 4 * hh + vv];
      }
      // two independent accumulator chains -> consecutive WMMAs have no RAW hazard
      acc0 = __builtin_amdgcn_wmma_f32_16x16x32_bf16(false, A0.v, false, Bk[kk].v,
                                                     (short)0, acc0, false, false);
      acc1 = __builtin_amdgcn_wmma_f32_16x16x32_bf16(false, A1.v, false, Bk[kk].v,
                                                     (short)0, acc1, false, false);
    }

    // commit pair p+1 (loaded last iteration) AFTER this pair's WMMAs are issued
    if (p + 1 < PTILES) commitP(cur ^ 1, ca);

    // C/D 16x16 f32 layout: col = lane&15, row = 8*half + r
    #pragma unroll
    for (int r = 0; r < 8; ++r) {
      int row0 = p * 32 + 8 * hh + r;
      int row1 = row0 + 16;
      if (row0 < NNODE) Whf[row0 * WHS + wave * 16 + l15] = acc0[r];
      if (row1 < NNODE) Whf[row1 * WHS + wave * 16 + l15] = acc1[r];
    }

    #pragma unroll
    for (int i = 0; i < 4; ++i) ca[i] = lb[i];   // retire pipeline registers
    __syncthreads();
  }

  // ---------------- attention logits ----------------
  float base = 0.f;                               // Wh[0,:] . a1 (LDS broadcast reads)
  #pragma unroll 8
  for (int o = 0; o < DOUT; ++o) base += Whf[o] * aLds[o];

  float e;
  if (t < NNODE) {
    float d = 0.f;
    #pragma unroll 8
    for (int o = 0; o < DOUT; ++o) d += Whf[t * WHS + o] * aLds[DOUT + o];
    e = d + base;
    e = (e > 0.f) ? e : NEGSL * e;                // LeakyReLU(0.2)
    if (adj[(size_t)b * NNODE + t] == 0) e = -__builtin_inff();
  } else {
    e = -__builtin_inff();
  }

  // ---------------- masked softmax (wave32 shuffles + 8-entry LDS combine) ----
  float wm = wave_max32(e);
  if (lane == 0) rbuf[wave] = wm;
  __syncthreads();
  float mx = rbuf[0];
  #pragma unroll
  for (int i = 1; i < 8; ++i) mx = fmaxf(mx, rbuf[i]);

  const float p  = (t < NNODE) ? __expf(e - mx) : 0.f;
  const float ws = wave_sum32(p);
  __syncthreads();                                // all rbuf max-reads done
  if (lane == 0) rbuf[wave] = ws;
  __syncthreads();
  float sum = rbuf[0];
  #pragma unroll
  for (int i = 1; i < 8; ++i) sum += rbuf[i];

  ebuf[t] = p * (1.f / sum);                      // attention weight for node t
  __syncthreads();

  // ---------------- h' = attn (1x200) @ Wh (200x128), all 256 threads ---------
  {
    const int o  = t & 127;
    const int n0 = (t >> 7) * 100;                // half-range per thread group
    float acc2 = 0.f;
    #pragma unroll 4
    for (int n = n0; n < n0 + 100; ++n) acc2 += ebuf[n] * Whf[n * WHS + o];
    rbuf[t] = acc2;
    __syncthreads();
    if (t < DOUT) out[(size_t)b * DOUT + t] = rbuf[t] + rbuf[t + 128];
  }
}

extern "C" void kernel_launch(void* const* d_in, const int* in_sizes, int n_in,
                              void* d_out, int out_size, void* d_ws, size_t ws_size,
                              hipStream_t stream) {
  // setup_inputs order: h (f32), adj_mask (int), W (f32), a (f32)
  const float* h   = (const float*)d_in[0];
  const int*   adj = (const int*)d_in[1];
  const float* W   = (const float*)d_in[2];
  const float* a   = (const float*)d_in[3];
  float* out = (float*)d_out;
  (void)in_sizes; (void)n_in; (void)out_size; (void)d_ws; (void)ws_size;

  gat_fused_kernel<<<BATCH, 256, 0, stream>>>(h, adj, W, a, out);
}
